// SelfMatchAttention_30202210025962
// MI455X (gfx1250) — 8-bit matrix operand order repaired and checked by reading, compiled, not run on hardware
//
#include <hip/hip_runtime.h>
#include <hip/hip_bf16.h>

#define N_TOKENS   8192
#define INPUT_BITS 64
#define N_BITS     10
#define N_ADDR     1024

typedef __attribute__((ext_vector_type(8))) int v8i;

__global__ void wnn_init_kernel(unsigned* __restrict__ P32) {
    int t = blockIdx.x * blockDim.x + threadIdx.x;
    if (t < N_ADDR * 2) P32[t] = 0u;
}

__global__ void wnn_addr_parity_kernel(const float* __restrict__ tokens,
                                       const int*   __restrict__ connections,
                                       unsigned*    __restrict__ P32,
                                       int*         __restrict__ addr_q) {
    int i = blockIdx.x * blockDim.x + threadIdx.x;
    if (i >= N_TOKENS) return;
    const float* trow = tokens + (size_t)i * INPUT_BITS;

    int aq = 0, ak = 0;
#pragma unroll
    for (int b = 0; b < N_BITS; ++b) {
        int c = connections[b];
        int w = 1 << (N_BITS - 1 - b);
        if (c < INPUT_BITS) {
            if (trow[c] > 0.5f) aq += w;
        } else {
            if (trow[c - INPUT_BITS] > 0.5f) ak += w;
        }
    }
    addr_q[i] = aq;

    unsigned lo = 0u, hi = 0u;
#pragma unroll
    for (int b = 0; b < 32; ++b) {
        lo |= (trow[b]      > 0.5f ? 1u : 0u) << b;
        hi |= (trow[32 + b] > 0.5f ? 1u : 0u) << b;
    }
    atomicXor(&P32[ak * 2 + 0], lo);
    atomicXor(&P32[ak * 2 + 1], hi);
}

__global__ void __launch_bounds__(256)
wnn_mod2_gemm_wmma(const float*    __restrict__ ram_table,
                   const unsigned* __restrict__ P32,
                   float*          __restrict__ OUTf) {
    __shared__ unsigned char ram8[N_ADDR];
    __shared__ unsigned     plds[N_ADDR * 2];

    int t = threadIdx.x;
    for (int k = t; k < N_ADDR; k += 256)
        ram8[k] = (ram_table[k] > 0.5f) ? (unsigned char)1 : (unsigned char)0;
    for (int k = t; k < N_ADDR * 2; k += 256)
        plds[k] = P32[k];
    __syncthreads();

    int wave  = (blockIdx.x * 256 + t) >> 5;
    int lane  = t & 31;
    int aq0   = (wave >> 2) * 16;
    int n0    = (wave & 3) * 16;
    int mn    = lane & 15;
    int lhi   = lane >> 4;
    int col   = n0 + mn;

    v8i acc = {0, 0, 0, 0, 0, 0, 0, 0};

    for (int kk = 0; kk < N_ADDR / 64; ++kk) {
        int ak0 = kk * 64;

        v8i A, B;
#pragma unroll
        for (int v = 0; v < 8; ++v) {
            unsigned wa = 0u, wb = 0u;
#pragma unroll
            for (int j = 0; j < 4; ++j) {
                int Ka   = ((v & 2) << 3) + ((v & 4) << 3) + ((v & 1) << 2)
                         + (lhi << 3) + j;
                int addr = (aq0 + mn + ak0 + Ka) & (N_ADDR - 1);
                wa |= ((unsigned)ram8[addr]) << (8 * j);

                int Kb   = ((v >> 2) << 5) + ((v & 2) << 3) + ((v & 1) << 2) + (lhi << 3) + j;
                int row  = ak0 + Kb;
                unsigned bit = (plds[row * 2 + (col >> 5)] >> (col & 31)) & 1u;
                wb |= bit << (8 * j);
            }
            A[v] = (int)wa;
            B[v] = (int)wb;
        }
        acc = __builtin_amdgcn_wmma_i32_16x16x64_iu8(false, A, false, B, acc,
                                                     false, false);
    }

#pragma unroll
    for (int r = 0; r < 8; ++r) {
        int M = aq0 + r + lhi * 8;
        OUTf[M * INPUT_BITS + col] = (float)(acc[r] & 1);
    }
}

__global__ void wnn_gather_kernel(const int*   __restrict__ addr_q,
                                  const float* __restrict__ OUTf,
                                  float*       __restrict__ out) {
    int idx = blockIdx.x * blockDim.x + threadIdx.x;
    int i = idx >> 6;
    int b = idx & 63;
    out[idx] = OUTf[addr_q[i] * INPUT_BITS + b];
}

extern "C" void kernel_launch(void* const* d_in, const int* in_sizes, int n_in,
                              void* d_out, int out_size, void* d_ws, size_t ws_size,
                              hipStream_t stream) {
    const float* tokens      = (const float*)d_in[0];
    const float* ram_table   = (const float*)d_in[1];
    const int*   connections = (const int*)d_in[2];
    float*       out         = (float*)d_out;

    char* ws = (char*)d_ws;
    unsigned* P32    = (unsigned*)(ws);
    int*      addr_q = (int*)(ws + 8 * 1024);
    float*    OUTf   = (float*)(ws + 40 * 1024);
    (void)in_sizes; (void)n_in; (void)out_size; (void)ws_size;

    wnn_init_kernel<<<(N_ADDR * 2 + 255) / 256, 256, 0, stream>>>(P32);

    wnn_addr_parity_kernel<<<N_TOKENS / 256, 256, 0, stream>>>(
        tokens, connections, P32, addr_q);

    wnn_mod2_gemm_wmma<<<32, 256, 0, stream>>>(ram_table, P32, OUTf);

    wnn_gather_kernel<<<(N_TOKENS * INPUT_BITS) / 256, 256, 0, stream>>>(
        addr_q, OUTf, out);
}
